// MemoryGraph_12223476924583
// MI455X (gfx1250) — matrix operand order repaired and checked by reading, not compiled, not run on hardware
//
#include <hip/hip_runtime.h>
#include <math.h>

typedef __bf16 bf16;
typedef unsigned int u32;
typedef __attribute__((ext_vector_type(16))) __bf16 v16bf;
typedef __attribute__((ext_vector_type(8)))  float  v8f;

#define NN 256   // N
#define DD 64    // D
#define HM 256   // Hm
#define NCC 64   // NC

__device__ __forceinline__ float sigm(float x) { return 1.0f / (1.0f + __expf(-x)); }

__device__ __forceinline__ v8f wmma_bf16(const v16bf& a, const v16bf& b, const v8f& c) {
    return __builtin_amdgcn_wmma_f32_16x16x32_bf16(false, a, false, b, (short)0, c, false, false);
}

union FragU { v16bf v; uint4 q[2]; };

// two 16-byte LDS loads -> one 16-element bf16 fragment
__device__ __forceinline__ v16bf ld2(const bf16* p0, const bf16* p1) {
    FragU f;
    f.q[0] = *(const uint4*)p0;
    f.q[1] = *(const uint4*)p1;
    return f.v;
}
__device__ __forceinline__ v16bf ld16(const bf16* p) { return ld2(p, p + 8); }

__device__ __forceinline__ u32 pack_bf16(float a, float b) {
    union { bf16 h[2]; u32 w; } u_;
    u_.h[0] = (bf16)a; u_.h[1] = (bf16)b;
    return u_.w;   // fuses to v_cvt_pk_bf16_f32
}

__global__ __launch_bounds__(256, 1) void memgraph_kernel(
    const float* __restrict__ xg,   const float* __restrict__ hg,
    const float* __restrict__ Wg,   const float* __restrict__ hebg,
    const float* __restrict__ nidg, const float* __restrict__ w1,
    const float* __restrict__ b1,   const float* __restrict__ w2,
    const float* __restrict__ b2,   const float* __restrict__ injWg,
    const float* __restrict__ injBg,const float* __restrict__ wdlg,
    const float* __restrict__ dglg, const float* __restrict__ hdlg,
    const int*   __restrict__ ipig, const int*   __restrict__ opig,
    float* __restrict__ outH, float* __restrict__ outW,
    float* __restrict__ outHeb, float* __restrict__ outR)
{
    // sPre is stored pre-swizzled in B-fragment order:
    //   slot = (((kt*4 + nt)*32) + lane)*16 + u
    // so each lane's B operand is 32 contiguous bytes (2 x ds_load_b128).
    __shared__ __align__(16) bf16  sPre[NN * DD];    // 32 KB
    __shared__ __align__(16) bf16  sHnew[NN * DD];   // 32 KB: agg bounce, then h_new
    __shared__ __align__(16) bf16  sHid[8][16 * 32]; // 8 KB: per-wave hidden bounce
    __shared__ float sInj[256];
    __shared__ float sRead[4 * DD];
    __shared__ float sXv[DD];
    __shared__ int   sIdx[8];

    const int bc = blockIdx.x;
    const int c  = bc & (NCC - 1);
    const int t  = threadIdx.x;
    const int wv = t >> 5, lane = t & 31, hl = lane >> 4, l16 = lane & 15;

    const float* x   = xg   + bc * DD;
    const float* h   = hg   + (size_t)bc * NN * DD;
    const float* W   = Wg   + (size_t)bc * NN * NN;
    const float* heb = hebg + (size_t)bc * NN * NN;
    const float* nid = nidg + (size_t)c * NN * DD;
    const float* iw  = injWg + (size_t)c * NN * DD;
    const float* ibv = injBg + c * NN;
    const float* wdl = wdlg + c * NN;
    const float* dgl = dglg + c * NN;
    const float* hdl = hdlg + c * NN;
    const int*   ipi = ipig + c * 4;
    const int*   opi = opig + c * 4;
    float* oh   = outH   + (size_t)bc * NN * DD;
    float* ow   = outW   + (size_t)bc * NN * NN;
    float* oheb = outHeb + (size_t)bc * NN * NN;
    float* orr  = outR   + (size_t)bc * 256;

    // ---------------- Phase A: pre -> swizzled LDS bf16; inj vector ----------
    if (t < DD) sXv[t] = x[t];
    if (t < 8)  sIdx[t] = (t < 4) ? ipi[t] : opi[t - 4];
    __syncthreads();
    {
        const int n    = t & 63;        // column d of pre
        const int kg   = t >> 6;        // row-pair group 0..3
        const int nt   = n >> 4;
        const int l16n = n & 15;
        #pragma unroll 4
        for (int m = 0; m < 32; ++m) {
            const int k0 = 2 * (m * 4 + kg);                  // even row
            const float a0 = h[k0 * DD + n]       + nid[k0 * DD + n];
            const float a1 = h[(k0 + 1) * DD + n] + nid[(k0 + 1) * DD + n];
            const int kt    = k0 >> 5;
            const int kin   = k0 & 31;
            const int lhalf = (kin >> 3) & 1;
            const int u     = (kin & 7) | ((kin >> 4) << 3);   // even
            const int ln32  = lhalf * 16 + l16n;
            const int slot  = (((kt * 4 + nt) * 32) + ln32) * 16 + u;
            ((u32*)sPre)[slot >> 1] = pack_bf16(a0, a1);
        }
        float s = ibv[t];
        #pragma unroll 8
        for (int k = 0; k < DD; ++k) s += sXv[k] * iw[t * DD + k];
        sInj[t] = s;
    }
    __syncthreads();

    // ---------------- Phase B: agg = W @ pre ---------------------------------
    #pragma unroll 1
    for (int mi = 0; mi < 2; ++mi) {
        const int mb = (2 * wv + mi) * 16;
        v8f acc[4] = {};
        const float* wr = W + (size_t)(mb + l16) * NN;
        #pragma unroll
        for (int kt = 0; kt < 8; ++kt) {
            const int kb = kt * 32 + hl * 8;
            const float4 f0 = *(const float4*)(wr + kb);
            const float4 f1 = *(const float4*)(wr + kb + 4);
            const float4 f2 = *(const float4*)(wr + kb + 16);
            const float4 f3 = *(const float4*)(wr + kb + 20);
            v16bf A;
            A[0]=(bf16)f0.x; A[1]=(bf16)f0.y; A[2]=(bf16)f0.z; A[3]=(bf16)f0.w;
            A[4]=(bf16)f1.x; A[5]=(bf16)f1.y; A[6]=(bf16)f1.z; A[7]=(bf16)f1.w;
            A[8]=(bf16)f2.x; A[9]=(bf16)f2.y; A[10]=(bf16)f2.z; A[11]=(bf16)f2.w;
            A[12]=(bf16)f3.x; A[13]=(bf16)f3.y; A[14]=(bf16)f3.z; A[15]=(bf16)f3.w;
            #pragma unroll
            for (int nt = 0; nt < 4; ++nt) {
                const v16bf Bf = ld16(sPre + (((kt * 4 + nt) * 32) + lane) * 16);
                acc[nt] = wmma_bf16(A, Bf, acc[nt]);
            }
        }
        #pragma unroll
        for (int nt = 0; nt < 4; ++nt)
            #pragma unroll
            for (int j = 0; j < 8; ++j)
                sHnew[(mb + j + 8 * hl) * DD + nt * 16 + l16] = (bf16)acc[nt][j];
    }

    // ---------------- Phase C: msg MLP + h_new (wave-local rows) --------------
    #pragma unroll 1
    for (int mi = 0; mi < 2; ++mi) {
        const int mb = (2 * wv + mi) * 16;
        v16bf Aag[2];
        #pragma unroll
        for (int k2 = 0; k2 < 2; ++k2) {
            const bf16* ar = &sHnew[(mb + l16) * DD + k2 * 32 + hl * 16];
            Aag[k2] = ld2(ar, ar + 8);
        }
        v8f msg[4] = {};
        #pragma unroll 1
        for (int hb = 0; hb < 8; ++hb) {
            #pragma unroll
            for (int hh = 0; hh < 2; ++hh) {
                v8f hid = {};
                #pragma unroll
                for (int k2 = 0; k2 < 2; ++k2) {
                    v16bf Bf;                        // B[k][n] = msg_w1[n][k]
                    const int n = hb * 32 + hh * 16 + l16;
                    const float* w1r = w1 + n * DD + k2 * 32 + hl * 16;
                    #pragma unroll
                    for (int u = 0; u < 16; ++u) Bf[u] = (bf16)w1r[u];
                    hid = wmma_bf16(Aag[k2], Bf, hid);
                }
                const float bb = b1[hb * 32 + hh * 16 + l16];
                #pragma unroll
                for (int j = 0; j < 8; ++j)
                    sHid[wv][(j + 8 * hl) * 32 + hh * 16 + l16] = (bf16)tanhf(hid[j] + bb);
            }
            v16bf A2;
            {
                const bf16* ar = &sHid[wv][l16 * 32 + hl * 16];
                A2 = ld2(ar, ar + 8);
            }
            #pragma unroll
            for (int nt = 0; nt < 4; ++nt) {
                v16bf B2;                            // B[k][n] = msg_w2[n][k]
                const int n = nt * 16 + l16;
                const float* w2r = w2 + n * HM + hb * 32 + hl * 16;
                #pragma unroll
                for (int u = 0; u < 16; ++u) B2[u] = (bf16)w2r[u];
                msg[nt] = wmma_bf16(A2, B2, msg[nt]);
            }
        }
        float g1m[8];
        #pragma unroll
        for (int j = 0; j < 8; ++j) g1m[j] = 1.0f - 0.5f * sigm(dgl[mb + j + 8 * hl]);
        #pragma unroll
        for (int nt = 0; nt < 4; ++nt) {
            const int col = nt * 16 + l16;
            const float bb2 = b2[col];
            #pragma unroll
            for (int j = 0; j < 8; ++j) {
                const int row = mb + j + 8 * hl;
                float v = msg[nt][j] + bb2;
                #pragma unroll
                for (int a = 0; a < 4; ++a) if (sIdx[a] == row) v += sInj[a * DD + col];
                const float hv = g1m[j] * h[row * DD + col] + tanhf(v);
                oh[row * DD + col] = hv;
                #pragma unroll
                for (int a = 0; a < 4; ++a) if (sIdx[4 + a] == row) sRead[a * DD + col] = hv;
                sHnew[row * DD + col] = (bf16)hv;
            }
        }
    }
    __syncthreads();

    // ---------------- Phase D: readout ---------------------------------------
    orr[t] = sRead[t] * 0.5f;                        // ALPHA^-0.5

    // ---------------- Phase E: outer, hebbian, W update + row RMS -------------
    #pragma unroll 1
    for (int mi = 0; mi < 2; ++mi) {
        const int mb = (2 * wv + mi) * 16;
        v8f acc[16];
        const v8f vz = {};
        #pragma unroll
        for (int i = 0; i < 16; ++i) acc[i] = vz;
        #pragma unroll
        for (int kt = 0; kt < 2; ++kt) {
            const bf16* ar = &sHnew[(mb + l16) * DD + kt * 32 + hl * 16];
            const v16bf A = ld2(ar, ar + 8);         // A[m][k] = h_new[row][d]
            const int kbb = kt * 32 + hl * 16;
            #pragma unroll
            for (int nt = 0; nt < 16; ++nt) {
                const v16bf Bf = ld16(&sHnew[(nt * 16 + l16) * DD + kbb]);
                acc[nt] = wmma_bf16(A, Bf, acc[nt]);
            }
        }
        float hg1m[8], wgv[8], wg1m[8];
        #pragma unroll
        for (int j = 0; j < 8; ++j) {
            const int row = mb + j + 8 * hl;
            const float hgj = 0.5f * sigm(hdl[row]);
            const float wgj = 0.5f * sigm(wdl[row]);
            hg1m[j] = 1.0f - hgj; wgv[j] = wgj; wg1m[j] = 1.0f - wgj;
        }
        v8f rss = vz;
        #pragma unroll
        for (int nt = 0; nt < 16; ++nt) {
            const int col = nt * 16 + l16;
            #pragma unroll
            for (int j = 0; j < 8; ++j) {
                const int row = mb + j + 8 * hl;
                const size_t off = (size_t)row * NN + col;
                const float outer = acc[nt][j] * (1.0f / 64.0f);
                const float hn = hg1m[j] * heb[off] + outer;
                oheb[off] = hn;
                float wu = wg1m[j] * W[off] + wgv[j] * hn;
                if (row == col) wu = 0.0f;
                acc[nt][j] = wu;
                rss[j] += wu * wu;
            }
        }
        float rn[8];
        #pragma unroll
        for (int j = 0; j < 8; ++j) {
            float s = rss[j];
            s += __shfl_xor(s, 1);
            s += __shfl_xor(s, 2);
            s += __shfl_xor(s, 4);
            s += __shfl_xor(s, 8);
            rn[j] = rsqrtf(s * (1.0f / 256.0f) + 1e-6f);
        }
        #pragma unroll
        for (int nt = 0; nt < 16; ++nt) {
            const int col = nt * 16 + l16;
            #pragma unroll
            for (int j = 0; j < 8; ++j) {
                const int row = mb + j + 8 * hl;
                ow[(size_t)row * NN + col] = acc[nt][j] * rn[j];
            }
        }
    }
}

extern "C" void kernel_launch(void* const* d_in, const int* in_sizes, int n_in,
                              void* d_out, int out_size, void* d_ws, size_t ws_size,
                              hipStream_t stream) {
    const float* x    = (const float*)d_in[0];
    const float* h    = (const float*)d_in[1];
    const float* W    = (const float*)d_in[2];
    const float* heb  = (const float*)d_in[3];
    const float* nid  = (const float*)d_in[4];
    const float* w1   = (const float*)d_in[5];
    const float* b1   = (const float*)d_in[6];
    const float* w2   = (const float*)d_in[7];
    const float* b2   = (const float*)d_in[8];
    const float* iw   = (const float*)d_in[9];
    const float* ib   = (const float*)d_in[10];
    const float* wdl  = (const float*)d_in[11];
    const float* dgl  = (const float*)d_in[12];
    const float* hdl  = (const float*)d_in[13];
    const int*   ipi  = (const int*)d_in[14];
    const int*   opi  = (const int*)d_in[15];

    float* outH   = (float*)d_out;
    float* outW   = outH   + (size_t)8 * 64 * 256 * 64;     // h_new
    float* outHeb = outW   + (size_t)8 * 64 * 256 * 256;    // W_new
    float* outR   = outHeb + (size_t)8 * 64 * 256 * 256;    // hebbian_new

    memgraph_kernel<<<dim3(512), dim3(256), 0, stream>>>(
        x, h, W, heb, nid, w1, b1, w2, b2, iw, ib, wdl, dgl, hdl, ipi, opi,
        outH, outW, outHeb, outR);
}
